// SimplifiedHAttention_27865747816878
// MI455X (gfx1250) — matrix operand order repaired and checked by reading, not compiled, not run on hardware
//
#include <hip/hip_runtime.h>
#include <hip/hip_bf16.h>

// Problem constants (from reference): B=2, S=2048, E=1024, H=16, dh=64
constexpr int BATCH = 2;
constexpr int SEQ   = 2048;
constexpr int EMB   = 1024;
constexpr int HEADS = 16;
constexpr int DH    = 64;
constexpr int MROWS = BATCH * SEQ;   // 4096

// LDS B-tile geometry: 64 rows (N) x 32 halfs (K), padded to 40 halfs/row
// (80B row stride -> 20-bank step, 16 consecutive rows hit 16 distinct banks)
constexpr int LDSROW = 40;

typedef __attribute__((ext_vector_type(16))) __bf16 v16bf;
typedef __attribute__((ext_vector_type(8)))  float  v8f;
typedef __attribute__((ext_vector_type(4)))  int    v4i;

union BF16Tile {
    unsigned short h[16];
    uint4          u4[2];
    v16bf          v;
};

__device__ __forceinline__ unsigned short f32_to_bf16(float f) {
    unsigned int u = __float_as_uint(f);
    unsigned int r = u + 0x7FFFu + ((u >> 16) & 1u);   // round-to-nearest-even
    return (unsigned short)(r >> 16);
}

// ---------------------------------------------------------------------------
// CDNA5 async global->LDS staging (ASYNCcnt path), with portable fallback.
// Builtin prototype: (global v4i*, local v4i*, imm offset, imm cpol)
// ---------------------------------------------------------------------------
#if defined(__has_builtin)
#  if __has_builtin(__builtin_amdgcn_global_load_async_to_lds_b128)
#    define USE_ASYNC_LDS 1
#  endif
#endif

typedef __attribute__((address_space(1))) v4i gv4i;
typedef __attribute__((address_space(3))) v4i lv4i;

__device__ __forceinline__ void lds_copy16(const unsigned short* src,
                                           unsigned short* dst) {
#if defined(USE_ASYNC_LDS)
    __builtin_amdgcn_global_load_async_to_lds_b128(
        (gv4i*)(unsigned short*)src, (lv4i*)dst, 0, 0);
#else
    *(uint4*)dst = *(const uint4*)src;
#endif
}

__device__ __forceinline__ void wait_async_zero() {
#if defined(USE_ASYNC_LDS)
#  if __has_builtin(__builtin_amdgcn_s_wait_asynccnt)
    __builtin_amdgcn_s_wait_asynccnt(0);
#  else
    asm volatile("s_wait_asynccnt 0" ::: "memory");
#  endif
#endif
}

// ---------------------------------------------------------------------------
// f32 -> bf16 conversion
// ---------------------------------------------------------------------------
__global__ __launch_bounds__(256) void cvt_bf16_kernel(
    const float* __restrict__ src, unsigned short* __restrict__ dst, int n) {
    int idx = blockIdx.x * blockDim.x + threadIdx.x;
    if (idx < n) dst[idx] = f32_to_bf16(src[idx]);
}

// ---------------------------------------------------------------------------
// WMMA GEMM:  C[m,n] = A[m,:] . W[n,:]  (+bias), bf16 in, f32 accum.
// Block = 4 waves = 64(M) x 64(N) macro-tile; B tile staged to LDS once per
// block per K-chunk (async, double-buffered); A double-buffered in registers.
// MODE 0: store bf16 to VT[b][n][s]  (transposed, for the attention stage)
// MODE 1: store f32  to out[m][n] + bias
// ---------------------------------------------------------------------------
template <int MODE>
__global__ __launch_bounds__(128) void gemm_wmma_kernel(
    const unsigned short* __restrict__ Amat,   // [MROWS][EMB] bf16
    const unsigned short* __restrict__ Wmat,   // [EMB][EMB]   bf16
    const float* __restrict__ bias,            // [EMB]
    void* __restrict__ outp) {
    const int lane = threadIdx.x & 31;
    const int wave = threadIdx.x >> 5;
    const int bm   = blockIdx.x >> 4;          // M/64 = 64
    const int bn   = blockIdx.x & 15;          // N/64 = 16
    const int m0   = bm * 64 + wave * 16;
    const int n0b  = bn * 64;
    const int hi   = lane >> 4;
    const int lr   = lane & 15;

    __shared__ unsigned short ldsB[2][64 * LDSROW];

    v8f acc[4] = {v8f{}, v8f{}, v8f{}, v8f{}};
    const unsigned short* aptr = Amat + (size_t)(m0 + lr) * EMB;

    auto stageB = [&](int buf, int kc) {
        const int k0 = kc * 32;
        #pragma unroll
        for (int u2 = 0; u2 < 2; ++u2) {
            const int u    = threadIdx.x * 2 + u2;   // 0..255 16B units
            const int row  = u >> 2;                 // 0..63
            const int unit = u & 3;                  // 0..3 (8 halfs each)
            lds_copy16(Wmat + (size_t)(n0b + row) * EMB + k0 + unit * 8,
                       &ldsB[buf][row * LDSROW + unit * 8]);
        }
    };
    auto loadA = [&](int kc) {
        const int k0 = kc * 32;
        BF16Tile A;
        A.u4[0] = *(const uint4*)(aptr + k0 + hi * 8);       // K chunks per ISA A layout
        A.u4[1] = *(const uint4*)(aptr + k0 + 16 + hi * 8);
        return A;
    };
    auto computeChunk = [&](int buf, const BF16Tile& A) {
        #pragma unroll
        for (int nt = 0; nt < 4; ++nt) {
            const unsigned short* bsrc = &ldsB[buf][(nt * 16 + lr) * LDSROW + hi * 8];
            BF16Tile Bt;
            Bt.u4[0] = ((const uint4*)bsrc)[0];
            Bt.u4[1] = ((const uint4*)bsrc)[2];
            acc[nt] = __builtin_amdgcn_wmma_f32_16x16x32_bf16(
                false, A.v, false, Bt.v, (short)0, acc[nt], false, false);
        }
    };

    constexpr int NK = EMB / 32;   // 32 (even)
    stageB(0, 0);
    BF16Tile Acur = loadA(0);
    wait_async_zero();
    __syncthreads();
    for (int kc = 0; kc < NK; kc += 2) {
        stageB(1, kc + 1);
        BF16Tile Anxt = loadA(kc + 1);
        computeChunk(0, Acur);
        wait_async_zero();
        __syncthreads();
        if (kc + 2 < NK) {
            stageB(0, kc + 2);
            Acur = loadA(kc + 2);
        }
        computeChunk(1, Anxt);
        if (kc + 2 < NK) wait_async_zero();
        __syncthreads();
    }

    if (MODE == 0) {
        // bf16, transposed: VT[b][n][s]
        unsigned short* VT = (unsigned short*)outp;
        const int bb = m0 >> 11;
        #pragma unroll
        for (int nt = 0; nt < 4; ++nt) {
            const int n = n0b + nt * 16 + lr;
            const float bs = bias[n];
            unsigned short* orow = VT + (size_t)(bb * EMB + n) * SEQ;
            #pragma unroll
            for (int r = 0; r < 8; ++r)
                orow[(m0 & 2047) + r + hi * 8] = f32_to_bf16(acc[nt][r] + bs);
        }
    } else {
        // f32, row-major: out[m][n]
        float* out = (float*)outp;
        #pragma unroll
        for (int nt = 0; nt < 4; ++nt) {
            const int n = n0b + nt * 16 + lr;
            const float bs = bias[n];
            #pragma unroll
            for (int r = 0; r < 8; ++r)
                out[(size_t)(m0 + r + hi * 8) * EMB + n] = acc[nt][r] + bs;
        }
    }
}

// ---------------------------------------------------------------------------
// Stage 2: hierarchical attention. Block = 4 waves sharing one (b,h); each
// wave owns a 16-row i-tile, all share the head's V^T tile staged in LDS.
// Weight tiles built branchlessly from lev in {0,1,2}:
//   bit1 = lev&1 (lev==1), bit2 = lev>>1 (lev==2)
//   w = (c0 + (c1-c0)*bit1 + (c2-c0)*bit2) * rrow * mask   (mask in {0,1})
// ---------------------------------------------------------------------------
__global__ __launch_bounds__(128) void attn_kernel(
    const int*   __restrict__ lev,            // [SEQ][SEQ]
    const int*   __restrict__ amask,          // [BATCH][SEQ]
    const float* __restrict__ hier,           // [MAXLEN][HEADS][3]
    const unsigned short* __restrict__ VT,    // [BATCH][EMB][SEQ] bf16
    unsigned short* __restrict__ Ab) {        // [MROWS][EMB] bf16 (b,s,e)
    const int lane = threadIdx.x & 31;
    const int wave = threadIdx.x >> 5;
    const int bh   = blockIdx.x >> 5;         // B*H = 32
    const int mg   = blockIdx.x & 31;         // S/64 = 32
    const int b    = bh >> 4;
    const int h    = bh & 15;
    const int m0   = (mg * 4 + wave) * 16;
    const int hi   = lane >> 4;
    const int lr   = lane & 15;

    __shared__ unsigned short ldsB[2][64 * LDSROW];

    const float c0 = hier[(b * HEADS + h) * 3 + 0];
    const float c1 = hier[(b * HEADS + h) * 3 + 1] * 0.5f;
    const float c2 = hier[(b * HEADS + h) * 3 + 2] * 0.25f;

    // rowsum for row i = m0+lr via integer level counting (branchless);
    // lane-half hi covers one j-half, halves combined by shfl.
    const int* lrow = lev + (size_t)(m0 + lr) * SEQ;
    int n1 = 0, n2 = 0;
    for (int j = hi * (SEQ / 2); j < (hi + 1) * (SEQ / 2); j += 4) {
        int4 l4 = *(const int4*)(lrow + j);
        n1 += (l4.x & 1) + (l4.y & 1) + (l4.z & 1) + (l4.w & 1);
        n2 += (l4.x >> 1) + (l4.y >> 1) + (l4.z >> 1) + (l4.w >> 1);
    }
    n1 += __shfl_xor(n1, 16, 32);
    n2 += __shfl_xor(n2, 16, 32);
    const float rowsum = c0 * (float)(SEQ - n1 - n2) + c1 * (float)n1 + c2 * (float)n2;
    const float rrow   = 1.0f / (rowsum + 1e-8f);
    const float a0 = c0 * rrow;
    const float a1 = (c1 - c0) * rrow;
    const float a2 = (c2 - c0) * rrow;
    const int*  mrow = amask + b * SEQ;

    v8f acc[4] = {v8f{}, v8f{}, v8f{}, v8f{}};

    auto stageB = [&](int buf, int kc) {       // V^T rows d=0..63 of this head
        const int j0 = kc * 32;
        #pragma unroll
        for (int u2 = 0; u2 < 2; ++u2) {
            const int u    = threadIdx.x * 2 + u2;
            const int row  = u >> 2;
            const int unit = u & 3;
            lds_copy16(VT + (size_t)(b * EMB + h * DH + row) * SEQ + j0 + unit * 8,
                       &ldsB[buf][row * LDSROW + unit * 8]);
        }
    };
    auto buildA = [&](int kc) {                // on-the-fly bf16 weight tile
        const int j0 = kc * 32;
        BF16Tile A;
        #pragma unroll
        for (int g = 0; g < 2; ++g) {
            const int jb = j0 + g * 16 + hi * 8;
            int4 l0 = *(const int4*)(lrow + jb);
            int4 l1 = *(const int4*)(lrow + jb + 4);
            int4 q0 = *(const int4*)(mrow + jb);
            int4 q1 = *(const int4*)(mrow + jb + 4);
            int lv[8] = {l0.x, l0.y, l0.z, l0.w, l1.x, l1.y, l1.z, l1.w};
            int mk[8] = {q0.x, q0.y, q0.z, q0.w, q1.x, q1.y, q1.z, q1.w};
            #pragma unroll
            for (int t = 0; t < 8; ++t) {
                float w = a0 + a1 * (float)(lv[t] & 1) + a2 * (float)(lv[t] >> 1);
                w *= (float)mk[t];             // mask is {0,1}
                A.h[g * 8 + t] = f32_to_bf16(w);
            }
        }
        return A;
    };
    auto computeChunk = [&](int buf, const BF16Tile& A) {
        #pragma unroll
        for (int nt = 0; nt < 4; ++nt) {
            const unsigned short* bsrc = &ldsB[buf][(nt * 16 + lr) * LDSROW + hi * 8];
            BF16Tile Bt;
            Bt.u4[0] = ((const uint4*)bsrc)[0];
            Bt.u4[1] = ((const uint4*)bsrc)[2];
            acc[nt] = __builtin_amdgcn_wmma_f32_16x16x32_bf16(
                false, A.v, false, Bt.v, (short)0, acc[nt], false, false);
        }
    };

    constexpr int NK = SEQ / 32;   // 64 (even)
    stageB(0, 0);
    BF16Tile Acur = buildA(0);
    wait_async_zero();
    __syncthreads();
    for (int kc = 0; kc < NK; kc += 2) {
        stageB(1, kc + 1);
        BF16Tile Anxt = buildA(kc + 1);
        computeChunk(0, Acur);
        wait_async_zero();
        __syncthreads();
        if (kc + 2 < NK) {
            stageB(0, kc + 2);
            Acur = buildA(kc + 2);
        }
        computeChunk(1, Anxt);
        if (kc + 2 < NK) wait_async_zero();
        __syncthreads();
    }

    // bf16 store into (b, s, e) layout for the output projection
    #pragma unroll
    for (int nt = 0; nt < 4; ++nt) {
        const int dd = h * DH + nt * 16 + lr;
        #pragma unroll
        for (int r = 0; r < 8; ++r) {
            const int irow = m0 + r + hi * 8;
            Ab[(size_t)(b * SEQ + irow) * EMB + dd] = f32_to_bf16(acc[nt][r]);
        }
    }
}

// ---------------------------------------------------------------------------
// Launch
// ---------------------------------------------------------------------------
extern "C" void kernel_launch(void* const* d_in, const int* in_sizes, int n_in,
                              void* d_out, int out_size, void* d_ws, size_t ws_size,
                              hipStream_t stream) {
    // setup_inputs order: x, attention_mask, level_indices, Wq, bq, Wk, bk,
    //                     Wv, bv, hier, Wo, bo   (Wq/bq/Wk/bk are dead code)
    const float* x    = (const float*)d_in[0];
    const int*   mask = (const int*)  d_in[1];
    const int*   lev  = (const int*)  d_in[2];
    const float* Wv   = (const float*)d_in[7];
    const float* bv   = (const float*)d_in[8];
    const float* hier = (const float*)d_in[9];
    const float* Wo   = (const float*)d_in[10];
    const float* bo   = (const float*)d_in[11];

    char* ws = (char*)d_ws;
    unsigned short* xb  = (unsigned short*)(ws);                        // 8 MB
    unsigned short* Wvb = (unsigned short*)(ws + (size_t)( 8 << 20));   // 2 MB
    unsigned short* Wob = (unsigned short*)(ws + (size_t)(10 << 20));   // 2 MB
    unsigned short* VT  = (unsigned short*)(ws + (size_t)(12 << 20));   // 8 MB
    unsigned short* Ab  = (unsigned short*)(ws + (size_t)(20 << 20));   // 8 MB

    const int nX = MROWS * EMB;   // 4194304
    const int nW = EMB * EMB;     // 1048576
    cvt_bf16_kernel<<<nX / 256, 256, 0, stream>>>(x,  xb,  nX);
    cvt_bf16_kernel<<<nW / 256, 256, 0, stream>>>(Wv, Wvb, nW);
    cvt_bf16_kernel<<<nW / 256, 256, 0, stream>>>(Wo, Wob, nW);

    gemm_wmma_kernel<0><<<1024, 128, 0, stream>>>(xb, Wvb, bv, (void*)VT);
    attn_kernel        <<<1024, 128, 0, stream>>>(lev, mask, hier, VT, Ab);
    gemm_wmma_kernel<1><<<1024, 128, 0, stream>>>(Ab, Wob, bo, d_out);
}
